// RTAGCNLayer_43473658970772
// MI455X (gfx1250) — matrix operand order repaired and checked by reading, not compiled, not run on hardware
//
#include <hip/hip_runtime.h>
#include <hip/hip_bf16.h>

typedef __bf16 bf16;
typedef bf16  v16bf __attribute__((ext_vector_type(16)));
typedef float v8f   __attribute__((ext_vector_type(8)));

#define DM 64

// ---------- helpers ----------

__device__ inline void atomicMaxF32(float* addr, float val) {
    unsigned int* ua = (unsigned int*)addr;
    unsigned int old = __float_as_uint(*((volatile float*)addr));
    while (__uint_as_float(old) < val) {
        unsigned int assumed = old;
        old = atomicCAS(ua, assumed, __float_as_uint(val));
        if (old == assumed) break;
    }
}

// Build a 16x32 bf16 A-fragment row slice for this lane.
// p points at the first 8-float run; second run is at p+16.
// (ISA 16-bit A layout: elems 0..7 = K kb..kb+7, elems 8..15 = K kb+16..kb+23)
__device__ inline v16bf make_frag(const float* p) {
    const float4* q = (const float4*)p;
    float4 r0 = q[0];   // p[0..3]
    float4 r1 = q[1];   // p[4..7]
    float4 r2 = q[4];   // p[16..19]
    float4 r3 = q[5];   // p[20..23]
    v16bf a;
    a[0]=(bf16)r0.x; a[1]=(bf16)r0.y; a[2]=(bf16)r0.z; a[3]=(bf16)r0.w;
    a[4]=(bf16)r1.x; a[5]=(bf16)r1.y; a[6]=(bf16)r1.z; a[7]=(bf16)r1.w;
    a[8]=(bf16)r2.x; a[9]=(bf16)r2.y; a[10]=(bf16)r2.z; a[11]=(bf16)r2.w;
    a[12]=(bf16)r3.x; a[13]=(bf16)r3.y; a[14]=(bf16)r3.z; a[15]=(bf16)r3.w;
    return a;
}

// ---------- kernel 0: init segMax/segSum/agg ----------

__global__ void init_kernel(float* __restrict__ segMax, float* __restrict__ segSum,
                            float* __restrict__ agg, int n, long aggTotal) {
    long i = (long)blockIdx.x * 256 + threadIdx.x;
    if (i < aggTotal) agg[i] = 0.0f;
    if (i < n) { segMax[i] = -3.0e38f; segSum[i] = 0.0f; }
}

// ---------- kernel 1: per-edge GEMMs via WMMA + attention score ----------
// One wave processes a tile of 16 edges. Weights staged in LDS as bf16 in
// WMMA-B fragment order: frag f = c*4+t, offset f*512 + lane*16 elements.

__global__ __launch_bounds__(256)
void edge_gemm_kernel(const float* __restrict__ h,
                      const float* __restrict__ eh,
                      const float* __restrict__ qrh,
                      const int*   __restrict__ src,
                      const int*   __restrict__ dst,
                      const float* __restrict__ Wmsg,   // [128,64] row-major (k,n)
                      const float* __restrict__ Wq,     // [64,64]
                      const float* __restrict__ Wk,     // [128,64]
                      float* __restrict__ msgOut,       // [E,64]
                      float* __restrict__ attOut,       // [E]
                      float* __restrict__ segMax,       // [N]
                      int nTiles) {
    __shared__ alignas(64) bf16 ldsW[20480];  // Wmsg 16 frags | Wk 16 frags | Wq 8 frags

    // cooperative fill: B layout -> lanes 0-15 hold K 32c+0..7,32c+16..23; lanes 16-31 hold K 32c+8..15,32c+24..31
    for (int idx = threadIdx.x; idx < 20480; idx += 256) {
        int f    = idx >> 9;
        int lane = (idx >> 4) & 31;
        int i    = idx & 15;
        const float* W; int c, t;
        if (f < 16)      { W = Wmsg; c = f >> 2;        t = f & 3; }
        else if (f < 32) { W = Wk;   c = (f - 16) >> 2; t = f & 3; }
        else             { W = Wq;   c = (f - 32) >> 2; t = f & 3; }
        int K  = 32 * c + ((lane < 16) ? 0 : 8) + ((i < 8) ? i : i + 8);
        int Nn = 16 * t + (lane & 15);
        ldsW[idx] = (bf16)W[K * DM + Nn];
    }
    __syncthreads();

    int wave = threadIdx.x >> 5;
    int lane = threadIdx.x & 31;
    int tile = blockIdx.x * 8 + wave;
    if (tile >= nTiles) return;

    int  mloc = lane & 15;
    long e    = (long)tile * 16 + mloc;
    int  s    = src[e];
    const float* hrow  = h   + (size_t)s * DM;
    const float* ehrow = eh  + (size_t)e * DM;
    const float* qrow  = qrh + (size_t)e * DM;
    int kb = (lane < 16) ? 0 : 8;

    v8f zero = {0.f,0.f,0.f,0.f,0.f,0.f,0.f,0.f};
    v8f accMsg[4], accK[4], accQ[4];
#pragma unroll
    for (int t = 0; t < 4; ++t) { accMsg[t] = zero; accK[t] = zero; accQ[t] = zero; }

    // cat = [h[src] | eh] : chunks 0,1 from h row, chunks 2,3 from eh row
#pragma unroll
    for (int c = 0; c < 4; ++c) {
        const float* p = (c < 2) ? (hrow + 32 * c + kb) : (ehrow + 32 * c - 64 + kb);
        v16bf a = make_frag(p);
#pragma unroll
        for (int t = 0; t < 4; ++t) {
            v16bf bM = *(const v16bf*)&ldsW[(c * 4 + t) * 512 + lane * 16];
            accMsg[t] = __builtin_amdgcn_wmma_f32_16x16x32_bf16(
                false, a, false, bM, (short)0, accMsg[t], false, false);
            v16bf bK = *(const v16bf*)&ldsW[8192 + (c * 4 + t) * 512 + lane * 16];
            accK[t] = __builtin_amdgcn_wmma_f32_16x16x32_bf16(
                false, a, false, bK, (short)0, accK[t], false, false);
        }
    }
    // q = qrh @ Wq (K = 64 -> 2 chunks)
#pragma unroll
    for (int c = 0; c < 2; ++c) {
        v16bf a = make_frag(qrow + 32 * c + kb);
#pragma unroll
        for (int t = 0; t < 4; ++t) {
            v16bf bQ = *(const v16bf*)&ldsW[16384 + (c * 4 + t) * 512 + lane * 16];
            accQ[t] = __builtin_amdgcn_wmma_f32_16x16x32_bf16(
                false, a, false, bQ, (short)0, accQ[t], false, false);
        }
    }

    // leaky_relu(msg) and store. C layout: VGPR r, lane -> (M = r + 8*(lane>=16), N = 16t + lane%16)
    long ebase   = (long)tile * 16;
    int  rowHalf = (lane >> 4) * 8;
#pragma unroll
    for (int t = 0; t < 4; ++t) {
#pragma unroll
        for (int r = 0; r < 8; ++r) {
            float v = accMsg[t][r];
            v = v > 0.f ? v : 0.01f * v;
            msgOut[(ebase + rowHalf + r) * DM + t * 16 + (lane & 15)] = v;
        }
    }

    // att[m] = (1/8) * sum_n q*k : per-lane partial over its column, reduce over 16 lanes
    float p8[8];
#pragma unroll
    for (int r = 0; r < 8; ++r) {
        float acc = 0.f;
#pragma unroll
        for (int t = 0; t < 4; ++t) acc += accQ[t][r] * accK[t][r];
        p8[r] = acc;
    }
#pragma unroll
    for (int m = 1; m < 16; m <<= 1) {
#pragma unroll
        for (int r = 0; r < 8; ++r) p8[r] += __shfl_xor(p8[r], m, 32);
    }
    // lane L picks row (L&7) of its half
    float attv = p8[0];
#pragma unroll
    for (int r = 1; r < 8; ++r) if ((lane & 7) == r) attv = p8[r];
    attv *= 0.125f;  // 1/sqrt(64)
    if ((lane & 8) == 0) {
        long ee = ebase + rowHalf + (lane & 7);
        attOut[ee] = attv;
        atomicMaxF32(&segMax[dst[ee]], attv);
    }
}

// ---------- kernel 2: e = exp(att - segMax[dst]); segSum += e ----------

__global__ void exp_kernel(float* __restrict__ att, const int* __restrict__ dst,
                           const float* __restrict__ segMax, float* __restrict__ segSum,
                           int E) {
    int e = blockIdx.x * 256 + threadIdx.x;
    if (e >= E) return;
    int d = dst[e];
    float ex = __expf(att[e] - segMax[d]);
    att[e] = ex;
    atomicAdd(&segSum[d], ex);
}

// ---------- kernel 3: agg[dst] += alpha * msg ----------

__global__ void scatter_kernel(const float* __restrict__ msg, const float* __restrict__ eatt,
                               const int* __restrict__ dst, const float* __restrict__ segSum,
                               float* __restrict__ agg, long total) {
    long gid = (long)blockIdx.x * 256 + threadIdx.x;
    if (gid >= total) return;
    long edge = gid >> 6;
    int  col  = (int)(gid & 63);
    int  d    = dst[edge];
    float alpha = eatt[edge] / segSum[d];
    atomicAdd(&agg[(long)d * DM + col], alpha * msg[gid]);
}

// ---------- kernel 4: x = agg + h; LayerNorm ----------

__global__ __launch_bounds__(256)
void ln_kernel(const float* __restrict__ agg, const float* __restrict__ h,
               const float* __restrict__ gamma, const float* __restrict__ beta,
               float* __restrict__ out, int N) {
    int wave = threadIdx.x >> 5;
    int lane = threadIdx.x & 31;
    int node = blockIdx.x * 8 + wave;
    if (node >= N) return;
    long b = (long)node * DM;
    float x0 = agg[b + lane]      + h[b + lane];
    float x1 = agg[b + lane + 32] + h[b + lane + 32];
    float s = x0 + x1;
#pragma unroll
    for (int m = 1; m < 32; m <<= 1) s += __shfl_xor(s, m, 32);
    float mu = s * (1.0f / 64.0f);
    float d0 = x0 - mu, d1 = x1 - mu;
    float v = d0 * d0 + d1 * d1;
#pragma unroll
    for (int m = 1; m < 32; m <<= 1) v += __shfl_xor(v, m, 32);
    float inv = rsqrtf(v * (1.0f / 64.0f) + 1e-6f);
    out[b + lane]      = d0 * inv * gamma[lane]      + beta[lane];
    out[b + lane + 32] = d1 * inv * gamma[lane + 32] + beta[lane + 32];
}

// ---------- launch ----------

extern "C" void kernel_launch(void* const* d_in, const int* in_sizes, int n_in,
                              void* d_out, int out_size, void* d_ws, size_t ws_size,
                              hipStream_t stream) {
    const float* h    = (const float*)d_in[0];
    const float* eh   = (const float*)d_in[1];
    const float* qrh  = (const float*)d_in[2];
    const int*   src  = (const int*)d_in[3];
    const int*   dst  = (const int*)d_in[4];
    const float* Wmsg = (const float*)d_in[5];
    const float* Wq   = (const float*)d_in[6];
    const float* Wk   = (const float*)d_in[7];
    const float* gam  = (const float*)d_in[8];
    const float* bet  = (const float*)d_in[9];

    int N = in_sizes[0] / DM;
    int E = in_sizes[1] / DM;

    float* ws     = (float*)d_ws;
    float* msg    = ws;                         // E*64
    float* att    = msg + (size_t)E * DM;       // E
    float* segMax = att + E;                    // N
    float* segSum = segMax + N;                 // N
    float* agg    = segSum + N;                 // N*64
    float* out    = (float*)d_out;

    long aggTotal = (long)N * DM;
    init_kernel<<<(int)((aggTotal + 255) / 256), 256, 0, stream>>>(segMax, segSum, agg, N, aggTotal);

    int nTiles = E / 16;  // E = 1,200,000 -> 75,000 tiles, 8 per block
    edge_gemm_kernel<<<(nTiles + 7) / 8, 256, 0, stream>>>(h, eh, qrh, src, dst,
                                                           Wmsg, Wq, Wk, msg, att, segMax, nTiles);

    exp_kernel<<<(E + 255) / 256, 256, 0, stream>>>(att, dst, segMax, segSum, E);

    long scTotal = (long)E * DM;
    scatter_kernel<<<(int)((scTotal + 255) / 256), 256, 0, stream>>>(msg, att, dst, segSum, agg, scTotal);

    ln_kernel<<<(N + 7) / 8, 256, 0, stream>>>(agg, h, gam, bet, out, N);
}
